// fTN_Transformer_Model_3178275799247
// MI455X (gfx1250) — matrix operand order repaired and checked by reading, not compiled, not run on hardware
//
#include <hip/hip_runtime.h>
#include <hip/hip_bf16.h>
#include <cstdint>
#include <cstddef>

#define DD   128
#define HH   8
#define LAY  6
#define NB   4          // batch
#define SSQ  100        // src len
#define TTQ  2048       // tgt len
#define SQRT_D 11.313708498984761f
#define LN10K  9.210340371976184f

#if __has_builtin(__builtin_amdgcn_global_load_async_to_lds_b128)
#define HAVE_ASYNC_LDS 1
#else
#define HAVE_ASYNC_LDS 0
#endif

typedef _Float16 half8 __attribute__((ext_vector_type(8)));
typedef _Float16 v16h  __attribute__((ext_vector_type(16)));
typedef _Float16 h4    __attribute__((ext_vector_type(4)));
typedef float    v8f   __attribute__((ext_vector_type(8)));
typedef float    f4    __attribute__((ext_vector_type(4)));
typedef int      vi4   __attribute__((vector_size(4 * sizeof(int))));   // matches builtin param

union AFrag { v16h v; half8 h[2]; _Float16 e[16]; };

// ---------------------------------------------------------------- positional enc
// faithful quirk: pe is indexed by the BATCH dim (x.shape[0] with batch-first)
__device__ __forceinline__ float pe_val(int b, int d) {
  int i2 = d & ~1;
  float div = __expf((float)i2 * (-LN10K / (float)DD));
  float ang = (float)b * div;
  return (d & 1) ? __cosf(ang) : __sinf(ang);
}

// ---------------------------------------------------------------- weight fp32 -> f16
__global__ void k_w2h(const float* __restrict__ w, _Float16* __restrict__ o, int n4) {
  int i = blockIdx.x * blockDim.x + threadIdx.x;
  if (i >= n4) return;
  f4 v = ((const f4*)w)[i];
  h4 h;
#pragma unroll
  for (int j = 0; j < 4; ++j) h[j] = (_Float16)v[j];
  ((h4*)o)[i] = h;
}

// ---------------------------------------------------------------- embeddings
__global__ void k_embed_src(const int* __restrict__ src, const float* __restrict__ emb,
                            _Float16* __restrict__ out) {
  int idx = blockIdx.x * blockDim.x + threadIdx.x;
  if (idx >= NB * SSQ * DD) return;
  int d = idx & (DD - 1);
  int bs = idx >> 7;
  int b = bs / SSQ;
  float v = emb[src[bs] * DD + d] * SQRT_D + pe_val(b, d);
  out[idx] = (_Float16)v;
}

__global__ void k_embed_tgt(const float* __restrict__ tgt, const float* __restrict__ fw,
                            const float* __restrict__ fb, _Float16* __restrict__ out) {
  int idx = blockIdx.x * blockDim.x + threadIdx.x;
  if (idx >= NB * TTQ * DD) return;
  int d = idx & (DD - 1);
  int row = idx >> 7;
  int b = row >> 11;               // TTQ = 2048
  float v = (tgt[row] * fw[d] + fb[d]) * SQRT_D + pe_val(b, d);
  out[idx] = (_Float16)v;
}

// ---------------------------------------------------------------- WMMA GEMM
// C[M,N] = act( A[M,K](f16) @ W[N,K]^T(f16) + bias ), C f16.
// One wave -> 32x64 tile: 8 WMMAs per k-step, B fragments reused across 2 A fragments.
__global__ __launch_bounds__(128)
void k_gemm(const _Float16* __restrict__ A, int lda,
            const _Float16* __restrict__ W, int ldw,
            const float* __restrict__ bias,
            _Float16* __restrict__ C, int ldc,
            int M, int N, int K, int relu) {
  int wid  = (blockIdx.x * blockDim.x + threadIdx.x) >> 5;
  int lane = threadIdx.x & 31;
  int mtiles = (M + 31) >> 5;
  int ntiles = N >> 6;
  if (wid >= mtiles * ntiles) return;           // wave-uniform exit (EXEC stays full)
  int mt = wid % mtiles, nt = wid / mtiles;
  int half_id = lane >> 4, l16 = lane & 15;

  int ar0 = mt * 32 + l16;      if (ar0 >= M) ar0 = M - 1;
  int ar1 = mt * 32 + 16 + l16; if (ar1 >= M) ar1 = M - 1;
  const _Float16* A0 = A + (size_t)ar0 * lda + half_id * 8;   // A kbase = 0 or 8
  const _Float16* A1 = A + (size_t)ar1 * lda + half_id * 8;
  int kb2 = half_id * 8;                                      // B kbase = 0 or 8

  v8f acc[2][4] = {};
  for (int k0 = 0; k0 < K; k0 += 32) {
    AFrag a0, a1;
    { const half8* p = (const half8*)(A0 + k0); a0.h[0] = p[0]; a0.h[1] = p[2]; }
    { const half8* p = (const half8*)(A1 + k0); a1.h[0] = p[0]; a1.h[1] = p[2]; }
    if (k0 + 32 < K) {
      __builtin_prefetch(A0 + k0 + 32, 0, 1);   // global_prefetch
    }
    AFrag bf[4];
#pragma unroll
    for (int j = 0; j < 4; ++j) {
      const half8* p = (const half8*)(W + (size_t)(nt * 64 + j * 16 + l16) * ldw + k0 + kb2);
      bf[j].h[0] = p[0];
      bf[j].h[1] = p[2];
    }
#pragma unroll
    for (int j = 0; j < 4; ++j) {
      acc[0][j] = __builtin_amdgcn_wmma_f32_16x16x32_f16(
          false, a0.v, false, bf[j].v, (short)0, acc[0][j], false, false);
      acc[1][j] = __builtin_amdgcn_wmma_f32_16x16x32_f16(
          false, a1.v, false, bf[j].v, (short)0, acc[1][j], false, false);
    }
  }
#pragma unroll
  for (int blk = 0; blk < 2; ++blk) {
#pragma unroll
    for (int j = 0; j < 4; ++j) {
      int n = nt * 64 + j * 16 + l16;
      float bv = bias ? bias[n] : 0.0f;
#pragma unroll
      for (int r = 0; r < 8; ++r) {
        int mg = mt * 32 + blk * 16 + r + half_id * 8;
        if (mg < M) {
          float v = acc[blk][j][r] + bv;
          if (relu) v = fmaxf(v, 0.0f);
          C[(size_t)mg * ldc + n] = (_Float16)v;
        }
      }
    }
  }
}

// ---------------------------------------------------------------- flash attention
// One wave per (b, h, 16-query tile); 32 keys per iteration.
// QK^T: 2 WMMAs (dh=16 zero-padded to 32); P@V: 1 WMMA with full K=32.
__global__ __launch_bounds__(128)
void k_attn(const _Float16* __restrict__ Q, int ldq,
            const _Float16* __restrict__ Kp, const _Float16* __restrict__ Vp, int ldkv,
            _Float16* __restrict__ O, int ldo, int Lq, int Lk) {
  __shared__ __attribute__((aligned(64))) _Float16 lds_p[4][16][32];
#if HAVE_ASYNC_LDS
  __shared__ __attribute__((aligned(64))) _Float16 lds_v[4][32][16];   // [key][dh]
#else
  __shared__ __attribute__((aligned(64))) _Float16 lds_vt[4][16][32];  // [dh][key]
#endif
  int ws   = threadIdx.x >> 5;
  int wid  = blockIdx.x * 4 + ws;
  int lane = threadIdx.x & 31;
  int nqt  = (Lq + 15) >> 4;
  if (wid >= NB * HH * nqt) return;
  int qt = wid % nqt;
  int h  = (wid / nqt) & (HH - 1);
  int b  = wid / (nqt * HH);
  int half_id = lane >> 4, l16 = lane & 15;
  half8 z8 = {};

  // Q A-fragment (loaded once): lanes 16-31 are K=16..31 -> zero padding
  int qrow = qt * 16 + l16; if (qrow >= Lq) qrow = Lq - 1;
  AFrag qf;
  {
    const half8* p = (const half8*)(Q + ((size_t)b * Lq + qrow) * ldq + h * 16);
    qf.h[0] = half_id ? z8 : p[0];
    qf.h[1] = half_id ? z8 : p[1];
  }

  float mrow[8], lrow[8];
  v8f acc = {};
#pragma unroll
  for (int r = 0; r < 8; ++r) { mrow[r] = -3.0e38f; lrow[r] = 0.0f; }

  int nkt = (Lk + 31) >> 5;
  for (int kt = 0; kt < nkt; ++kt) {
    int key0 = kt * 32;
    // two K B-fragments (16 keys each); lanes 16-31 zero (dh padding)
    int krA = key0 + l16;      if (krA >= Lk) krA = Lk - 1;
    int krB = key0 + 16 + l16; if (krB >= Lk) krB = Lk - 1;
    AFrag kfA, kfB;
    {
      const half8* p = (const half8*)(Kp + ((size_t)b * Lk + krA) * ldkv + h * 16);
      kfA.h[0] = half_id ? z8 : p[0];
      kfA.h[1] = half_id ? z8 : p[1];
    }
    {
      const half8* p = (const half8*)(Kp + ((size_t)b * Lk + krB) * ldkv + h * 16);
      kfB.h[0] = half_id ? z8 : p[0];
      kfB.h[1] = half_id ? z8 : p[1];
    }

    // stage V tile (32 keys x 16 dh) into LDS; one key-row per lane
    int kv = key0 + lane; if (kv >= Lk) kv = Lk - 1;
#if HAVE_ASYNC_LDS
    {
      // prior-iteration ds reads of lds_v must retire before the async unit
      // overwrites it (ASYNCcnt is unordered vs DScnt)
      asm volatile("s_wait_dscnt 0" ::: "memory");
      vi4* gsrc = (vi4*)(Vp + ((size_t)b * Lk + kv) * ldkv + h * 16);
      vi4* ldst = (vi4*)&lds_v[ws][lane][0];
      __builtin_amdgcn_global_load_async_to_lds_b128(gsrc, ldst, 0, 0);
      __builtin_amdgcn_global_load_async_to_lds_b128(gsrc, ldst, 16, 0);
    }
#else
    {
      const half8* pv = (const half8*)(Vp + ((size_t)b * Lk + kv) * ldkv + h * 16);
      half8 v0 = pv[0], v1 = pv[1];
#pragma unroll
      for (int i = 0; i < 8; ++i) {
        lds_vt[ws][i][lane]     = v0[i];
        lds_vt[ws][8 + i][lane] = v1[i];
      }
    }
#endif

    v8f sc0 = {}, sc1 = {};
    sc0 = __builtin_amdgcn_wmma_f32_16x16x32_f16(
        false, qf.v, false, kfA.v, (short)0, sc0, false, false);
    sc1 = __builtin_amdgcn_wmma_f32_16x16x32_f16(
        false, qf.v, false, kfB.v, (short)0, sc1, false, false);

    // online softmax over 32 columns; row m = r + 8*half_id lives in one
    // 16-lane half, so xor masks 1/2/4/8 are exact row reductions.
    bool inv0 = (key0 + l16) >= Lk;
    bool inv1 = (key0 + 16 + l16) >= Lk;
#pragma unroll
    for (int r = 0; r < 8; ++r) {
      float s0 = sc0[r] * 0.25f;             // 1/sqrt(dh)
      float s1 = sc1[r] * 0.25f;
      if (inv0) s0 = -3.0e38f;
      if (inv1) s1 = -3.0e38f;
      float mx = fmaxf(s0, s1);
      mx = fmaxf(mx, __shfl_xor(mx, 1));
      mx = fmaxf(mx, __shfl_xor(mx, 2));
      mx = fmaxf(mx, __shfl_xor(mx, 4));
      mx = fmaxf(mx, __shfl_xor(mx, 8));
      float mnew = fmaxf(mrow[r], mx);
      float corr = __expf(mrow[r] - mnew);
      float p0   = __expf(s0 - mnew);
      float p1   = __expf(s1 - mnew);
      float sum  = p0 + p1;
      sum += __shfl_xor(sum, 1);
      sum += __shfl_xor(sum, 2);
      sum += __shfl_xor(sum, 4);
      sum += __shfl_xor(sum, 8);
      lrow[r] = lrow[r] * corr + sum;
      mrow[r] = mnew;
      acc[r] *= corr;
      lds_p[ws][r + half_id * 8][l16]      = (_Float16)p0;
      lds_p[ws][r + half_id * 8][16 + l16] = (_Float16)p1;
    }
    __builtin_amdgcn_wave_barrier();
#if HAVE_ASYNC_LDS
    asm volatile("s_wait_asynccnt 0" ::: "memory");
#endif

    // P A-fragment, full K=32: kb=16*half, halves 0..7 = keys kb..kb+7, 8..15 = kb+8..kb+15
    AFrag pf;
    {
      const half8* pp0 = (const half8*)&lds_p[ws][l16][half_id * 16];
      const half8* pp1 = (const half8*)&lds_p[ws][l16][half_id * 16 + 8];
      pf.h[0] = pp0[0];
      pf.h[1] = pp1[0];
    }
    // V B-fragment, full K=32: lane n=l16 needs V[kb2+i][n]
    AFrag vf;
#if HAVE_ASYNC_LDS
    {
      int kb2v = half_id * 16;
#pragma unroll
      for (int i = 0; i < 16; ++i) vf.e[i] = lds_v[ws][kb2v + i][l16];
    }
#else
    {
      const half8* pv = (const half8*)&lds_vt[ws][l16][half_id * 16];
      vf.h[0] = pv[0];
      vf.h[1] = pv[1];
    }
#endif
    acc = __builtin_amdgcn_wmma_f32_16x16x32_f16(
        false, pf.v, false, vf.v, (short)0, acc, false, false);
  }

#pragma unroll
  for (int r = 0; r < 8; ++r) {
    int mg = qt * 16 + r + half_id * 8;
    if (mg < Lq) {
      float v = acc[r] / lrow[r];
      O[((size_t)b * Lq + mg) * ldo + h * 16 + l16] = (_Float16)v;
    }
  }
}

// ---------------------------------------------------------------- layernorm (+residual)
__global__ __launch_bounds__(128)
void k_ln(const _Float16* __restrict__ X, const _Float16* __restrict__ R,
          const float* __restrict__ g, const float* __restrict__ bta,
          _Float16* __restrict__ Y, int M) {
  int wid  = (blockIdx.x * blockDim.x + threadIdx.x) >> 5;
  int lane = threadIdx.x & 31;
  if (wid >= M) return;
  h4 xv = *(const h4*)(X + (size_t)wid * DD + lane * 4);
  float x[4];
#pragma unroll
  for (int i = 0; i < 4; ++i) x[i] = (float)xv[i];
  if (R) {
    h4 rv = *(const h4*)(R + (size_t)wid * DD + lane * 4);
#pragma unroll
    for (int i = 0; i < 4; ++i) x[i] += (float)rv[i];
  }
  float s = x[0] + x[1] + x[2] + x[3];
  s += __shfl_xor(s, 1);  s += __shfl_xor(s, 2);  s += __shfl_xor(s, 4);
  s += __shfl_xor(s, 8);  s += __shfl_xor(s, 16);
  float mean = s * (1.0f / DD);
  float vsum = 0.0f;
#pragma unroll
  for (int i = 0; i < 4; ++i) { float d = x[i] - mean; vsum += d * d; }
  vsum += __shfl_xor(vsum, 1);  vsum += __shfl_xor(vsum, 2);  vsum += __shfl_xor(vsum, 4);
  vsum += __shfl_xor(vsum, 8);  vsum += __shfl_xor(vsum, 16);
  float inv = 1.0f / sqrtf(vsum * (1.0f / DD) + 1e-5f);
  h4 yv;
#pragma unroll
  for (int i = 0; i < 4; ++i) {
    int d = lane * 4 + i;
    yv[i] = (_Float16)((x[i] - mean) * inv * g[d] + bta[d]);
  }
  *(h4*)(Y + (size_t)wid * DD + lane * 4) = yv;
}

// ---------------------------------------------------------------- final projection
__global__ __launch_bounds__(128)
void k_fc(const _Float16* __restrict__ Y, const float* __restrict__ fw,
          const float* __restrict__ fb, const float* __restrict__ tgt,
          float* __restrict__ out, int M) {
  int wid  = (blockIdx.x * blockDim.x + threadIdx.x) >> 5;
  int lane = threadIdx.x & 31;
  if (wid >= M) return;
  h4 yv = *(const h4*)(Y + (size_t)wid * DD + lane * 4);
  f4 wv = *(const f4*)(fw + lane * 4);
  float s = 0.0f;
#pragma unroll
  for (int i = 0; i < 4; ++i) s += (float)yv[i] * wv[i];
  s += __shfl_xor(s, 1);  s += __shfl_xor(s, 2);  s += __shfl_xor(s, 4);
  s += __shfl_xor(s, 8);  s += __shfl_xor(s, 16);
  if (lane == 0) out[wid] = tgt[wid] + 1e-3f * (s + fb[0]);
}

// ---------------------------------------------------------------- host orchestration
static inline void launch_gemm(const _Float16* A, int lda, const _Float16* W, int ldw,
                               const float* bias, _Float16* C, int ldc,
                               int M, int N, int K, int relu, hipStream_t s) {
  int mt = (M + 31) / 32, nt = N / 64;
  int waves = mt * nt;
  k_gemm<<<(waves + 3) / 4, 128, 0, s>>>(A, lda, W, ldw, bias, C, ldc, M, N, K, relu);
}
static inline void launch_attn(const _Float16* Q, int ldq, const _Float16* Kp,
                               const _Float16* Vp, int ldkv, _Float16* O, int ldo,
                               int Lq, int Lk, hipStream_t s) {
  int waves = NB * HH * ((Lq + 15) / 16);
  k_attn<<<(waves + 3) / 4, 128, 0, s>>>(Q, ldq, Kp, Vp, ldkv, O, ldo, Lq, Lk);
}
static inline void launch_ln(const _Float16* X, const _Float16* R, const float* g,
                             const float* b, _Float16* Y, int M, hipStream_t s) {
  k_ln<<<(M + 3) / 4, 128, 0, s>>>(X, R, g, b, Y, M);
}
static inline void launch_w2h(const float* w, _Float16* o, size_t n, hipStream_t s) {
  int n4 = (int)(n / 4);
  k_w2h<<<(n4 + 255) / 256, 256, 0, s>>>(w, o, n4);
}

extern "C" void kernel_launch(void* const* d_in, const int* in_sizes, int n_in,
                              void* d_out, int out_size, void* d_ws, size_t ws_size,
                              hipStream_t stream) {
  const int*   src     = (const int*)  d_in[0];
  const float* tgt     = (const float*)d_in[1];
  const float* emb     = (const float*)d_in[2];
  const float* fembw   = (const float*)d_in[3];
  const float* fembb   = (const float*)d_in[4];
  const float* enc_inw = (const float*)d_in[5];
  const float* enc_inb = (const float*)d_in[6];
  const float* enc_ow  = (const float*)d_in[7];
  const float* enc_ob  = (const float*)d_in[8];
  const float* enc_l1g = (const float*)d_in[9];
  const float* enc_l1b = (const float*)d_in[10];
  const float* enc_l2g = (const float*)d_in[11];
  const float* enc_l2b = (const float*)d_in[12];
  const float* enc_f1w = (const float*)d_in[13];
  const float* enc_f1b = (const float*)d_in[14];
  const float* enc_f2w = (const float*)d_in[15];
  const float* enc_f2b = (const float*)d_in[16];
  const float* enc_ng  = (const float*)d_in[17];
  const float* enc_nb  = (const float*)d_in[18];
  const float* dec_inw = (const float*)d_in[19];
  const float* dec_inb = (const float*)d_in[20];
  const float* dec_ow  = (const float*)d_in[21];
  const float* dec_ob  = (const float*)d_in[22];
  const float* dec_cinw= (const float*)d_in[23];
  const float* dec_cinb= (const float*)d_in[24];
  const float* dec_cow = (const float*)d_in[25];
  const float* dec_cob = (const float*)d_in[26];
  const float* dec_l1g = (const float*)d_in[27];
  const float* dec_l1b = (const float*)d_in[28];
  const float* dec_l2g = (const float*)d_in[29];
  const float* dec_l2b = (const float*)d_in[30];
  const float* dec_l3g = (const float*)d_in[31];
  const float* dec_l3b = (const float*)d_in[32];
  const float* dec_f1w = (const float*)d_in[33];
  const float* dec_f1b = (const float*)d_in[34];
  const float* dec_f2w = (const float*)d_in[35];
  const float* dec_f2b = (const float*)d_in[36];
  const float* dec_ng  = (const float*)d_in[37];
  const float* dec_nb  = (const float*)d_in[38];
  const float* fc_w    = (const float*)d_in[39];
  const float* fc_b    = (const float*)d_in[40];

  // ---- carve f16 workspace (~35 MB, 256B-aligned slices) ----
  char* wp = (char*)d_ws;
  auto carve = [&](size_t n) -> _Float16* {
    _Float16* p = (_Float16*)wp;
    wp += (n * sizeof(_Float16) + 255) & ~(size_t)255;
    return p;
  };
  const int ME = NB * SSQ;     // 400
  const int MD = NB * TTQ;     // 8192
  // f16 weight shadows (converted once per launch)
  const size_t sz_inw = (size_t)LAY * 3 * DD * DD;
  const size_t sz_ow  = (size_t)LAY * DD * DD;
  const size_t sz_f1  = (size_t)LAY * 512 * DD;
  const size_t sz_f2  = (size_t)LAY * DD * 512;
  _Float16* h_enc_inw  = carve(sz_inw);
  _Float16* h_enc_ow   = carve(sz_ow);
  _Float16* h_enc_f1w  = carve(sz_f1);
  _Float16* h_enc_f2w  = carve(sz_f2);
  _Float16* h_dec_inw  = carve(sz_inw);
  _Float16* h_dec_ow   = carve(sz_ow);
  _Float16* h_dec_cinw = carve(sz_inw);
  _Float16* h_dec_cow  = carve(sz_ow);
  _Float16* h_dec_f1w  = carve(sz_f1);
  _Float16* h_dec_f2w  = carve(sz_f2);
  // activations
  _Float16* memA = carve((size_t)ME * DD);
  _Float16* memB = carve((size_t)ME * DD);
  _Float16* memN = carve((size_t)ME * DD);
  _Float16* qkvE = carve((size_t)ME * 3 * DD);
  _Float16* oE   = carve((size_t)ME * DD);
  _Float16* tE   = carve((size_t)ME * DD);
  _Float16* hE   = carve((size_t)ME * 512);
  _Float16* yA   = carve((size_t)MD * DD);
  _Float16* yB   = carve((size_t)MD * DD);
  _Float16* yN   = carve((size_t)MD * DD);
  _Float16* qkvD = carve((size_t)MD * 3 * DD);
  _Float16* oD   = carve((size_t)MD * DD);
  _Float16* tD   = carve((size_t)MD * DD);
  _Float16* hD   = carve((size_t)MD * 512);
  _Float16* qD   = carve((size_t)MD * DD);
  _Float16* kvD  = carve((size_t)ME * 2 * DD);
  (void)ws_size; (void)n_in; (void)in_sizes; (void)out_size;

  // ---- convert weights to f16 once ----
  launch_w2h(enc_inw,  h_enc_inw,  sz_inw, stream);
  launch_w2h(enc_ow,   h_enc_ow,   sz_ow,  stream);
  launch_w2h(enc_f1w,  h_enc_f1w,  sz_f1,  stream);
  launch_w2h(enc_f2w,  h_enc_f2w,  sz_f2,  stream);
  launch_w2h(dec_inw,  h_dec_inw,  sz_inw, stream);
  launch_w2h(dec_ow,   h_dec_ow,   sz_ow,  stream);
  launch_w2h(dec_cinw, h_dec_cinw, sz_inw, stream);
  launch_w2h(dec_cow,  h_dec_cow,  sz_ow,  stream);
  launch_w2h(dec_f1w,  h_dec_f1w,  sz_f1,  stream);
  launch_w2h(dec_f2w,  h_dec_f2w,  sz_f2,  stream);

  // ---- embeddings + positional encoding ----
  k_embed_src<<<(ME * DD + 255) / 256, 256, 0, stream>>>(src, emb, memA);
  k_embed_tgt<<<(MD * DD + 255) / 256, 256, 0, stream>>>(tgt, fembw, fembb, yA);

  // ---- encoder ----
  _Float16 *mc = memA, *ma = memB;
  for (int i = 0; i < LAY; ++i) {
    launch_gemm(mc, DD, h_enc_inw + (size_t)i * 3 * DD * DD, DD, enc_inb + i * 3 * DD,
                qkvE, 3 * DD, ME, 3 * DD, DD, 0, stream);
    launch_attn(qkvE, 3 * DD, qkvE + DD, qkvE + 2 * DD, 3 * DD, oE, DD, SSQ, SSQ, stream);
    launch_gemm(oE, DD, h_enc_ow + (size_t)i * DD * DD, DD, enc_ob + i * DD,
                tE, DD, ME, DD, DD, 0, stream);
    launch_ln(mc, tE, enc_l1g + i * DD, enc_l1b + i * DD, ma, ME, stream);
    { _Float16* t = mc; mc = ma; ma = t; }
    launch_gemm(mc, DD, h_enc_f1w + (size_t)i * 512 * DD, DD, enc_f1b + i * 512,
                hE, 512, ME, 512, DD, 1, stream);
    launch_gemm(hE, 512, h_enc_f2w + (size_t)i * DD * 512, 512, enc_f2b + i * DD,
                tE, DD, ME, DD, 512, 0, stream);
    launch_ln(mc, tE, enc_l2g + i * DD, enc_l2b + i * DD, ma, ME, stream);
    { _Float16* t = mc; mc = ma; ma = t; }
  }
  launch_ln(mc, nullptr, enc_ng, enc_nb, memN, ME, stream);

  // ---- decoder ----
  _Float16 *yc = yA, *ya = yB;
  for (int i = 0; i < LAY; ++i) {
    // self attention
    launch_gemm(yc, DD, h_dec_inw + (size_t)i * 3 * DD * DD, DD, dec_inb + i * 3 * DD,
                qkvD, 3 * DD, MD, 3 * DD, DD, 0, stream);
    launch_attn(qkvD, 3 * DD, qkvD + DD, qkvD + 2 * DD, 3 * DD, oD, DD, TTQ, TTQ, stream);
    launch_gemm(oD, DD, h_dec_ow + (size_t)i * DD * DD, DD, dec_ob + i * DD,
                tD, DD, MD, DD, DD, 0, stream);
    launch_ln(yc, tD, dec_l1g + i * DD, dec_l1b + i * DD, ya, MD, stream);
    { _Float16* t = yc; yc = ya; ya = t; }
    // cross attention (q from y, k/v from encoder memory)
    launch_gemm(yc, DD, h_dec_cinw + (size_t)i * 3 * DD * DD, DD, dec_cinb + i * 3 * DD,
                qD, DD, MD, DD, DD, 0, stream);
    launch_gemm(memN, DD, h_dec_cinw + (size_t)i * 3 * DD * DD + DD * DD, DD,
                dec_cinb + i * 3 * DD + DD, kvD, 2 * DD, ME, 2 * DD, DD, 0, stream);
    launch_attn(qD, DD, kvD, kvD + DD, 2 * DD, oD, DD, TTQ, SSQ, stream);
    launch_gemm(oD, DD, h_dec_cow + (size_t)i * DD * DD, DD, dec_cob + i * DD,
                tD, DD, MD, DD, DD, 0, stream);
    launch_ln(yc, tD, dec_l2g + i * DD, dec_l2b + i * DD, ya, MD, stream);
    { _Float16* t = yc; yc = ya; ya = t; }
    // FFN
    launch_gemm(yc, DD, h_dec_f1w + (size_t)i * 512 * DD, DD, dec_f1b + i * 512,
                hD, 512, MD, 512, DD, 1, stream);
    launch_gemm(hD, 512, h_dec_f2w + (size_t)i * DD * 512, 512, dec_f2b + i * DD,
                tD, DD, MD, DD, 512, 0, stream);
    launch_ln(yc, tD, dec_l3g + i * DD, dec_l3b + i * DD, ya, MD, stream);
    { _Float16* t = yc; yc = ya; ya = t; }
  }
  launch_ln(yc, nullptr, dec_ng, dec_nb, yN, MD, stream);

  // ---- final head: out = tgt + ETA * (yN @ fc_w^T + fc_b) ----
  k_fc<<<(MD + 3) / 4, 128, 0, stream>>>(yN, fc_w, fc_b, tgt, (float*)d_out, MD);
}
